// Net_12180527251657
// MI455X (gfx1250) — matrix operand order repaired and checked by reading, not compiled, not run on hardware
//
#include <hip/hip_runtime.h>
#include <hip/hip_bf16.h>
#include <math.h>

typedef __attribute__((ext_vector_type(16))) _Float16 v16h;
typedef __attribute__((ext_vector_type(8)))  float    v8f;
typedef __attribute__((ext_vector_type(4)))  unsigned int u32x4;

#define T_STEPS 192
#define IN_DIM  24
#define H_DIM   64
#define KQ      20

// ---------------- fast transcendental helpers (v_exp/v_log/v_rcp) ----------------

__device__ __forceinline__ float sig_fast(float x) {
    return __builtin_amdgcn_rcpf(1.0f + __expf(-x));
}
__device__ __forceinline__ float tanh_fast(float x) {
    // 1 - 2/(e^{2x}+1): exact at +-inf, hardware exp+rcp
    return 1.0f - 2.0f * __builtin_amdgcn_rcpf(__expf(2.0f * x) + 1.0f);
}
__device__ __forceinline__ float softplus_fast(float x) {
    return fmaxf(x, 0.0f) + __logf(1.0f + __expf(-fabsf(x)));
}

// A-fragment (16x32 f16), contiguous k order matching load_b_frag:
// lane l<16: row=l, k= kOff+0..15 (2 x b128)
// lane l>=16: row=l-16, k= kOff+16..31
__device__ __forceinline__ v16h load_a_frag(const _Float16* base, int strideH, int kOff) {
    int lane = threadIdx.x & 31;
    int r = lane & 15;
    int kb = kOff + ((lane & 16) ? 16 : 0);
    union { v16h v; u32x4 q[2]; } u;
    const _Float16* p = base + r * strideH + kb;
    u.q[0] = *(const u32x4*)(p);
    u.q[1] = *(const u32x4*)(p + 8);
    return u.v;
}

// B-fragment (32x16 f16), k-major source Bm[k*N + col]:
// lane n<16: col=n, k = kOff+0..15 packed 2/VGPR; lane n>=16: col=n-16, k=kOff+16..31
__device__ __forceinline__ v16h load_b_frag(const _Float16* Bm, int N, int colBase, int kOff) {
    int lane = threadIdx.x & 31;
    int col = colBase + (lane & 15);
    int kb = kOff + ((lane & 16) ? 16 : 0);
    union { v16h v; _Float16 h[16]; } u;
#pragma unroll
    for (int j = 0; j < 8; ++j) {
        u.h[2 * j]     = Bm[(kb + 2 * j) * N + col];
        u.h[2 * j + 1] = Bm[(kb + 2 * j + 1) * N + col];
    }
    return u.v;
}

__device__ __forceinline__ v8f wmma_f16(v16h a, v16h b, v8f c) {
    return __builtin_amdgcn_wmma_f32_16x16x32_f16(false, a, false, b, (short)0, c, false, false);
}

// D (16x16 f32): VGPR r, lane n<16 -> D[r][n]; lane n+16 -> D[8+r][n]
__device__ __forceinline__ void store_d(float* dst, int strideF, int colBase, v8f acc) {
    int lane = threadIdx.x & 31;
    int rb = (lane & 16) ? 8 : 0;
    int col = colBase + (lane & 15);
#pragma unroll
    for (int r = 0; r < 8; ++r) dst[(rb + r) * strideF + col] = acc[r];
}

// ---------------- weight prep ----------------
// wsH layout (halves): Bih0[32][256] @0, Bhh0[64][256] @8192, Bih1[64][256] @24576,
//                      Bhh1[64][256] @40960, Bhead[128][32] @57344  (total 61440)
// wsF layout (floats): bias0[256], bias1[256], biasH[32]
__global__ void prep_kernel(const float* __restrict__ w_ih0, const float* __restrict__ w_hh0,
                            const float* __restrict__ b_ih0, const float* __restrict__ b_hh0,
                            const float* __restrict__ w_ih1, const float* __restrict__ w_hh1,
                            const float* __restrict__ b_ih1, const float* __restrict__ b_hh1,
                            const float* __restrict__ w_b0, const float* __restrict__ bb_b0,
                            const float* __restrict__ w_g,  const float* __restrict__ b_g,
                            _Float16* __restrict__ wsH, float* __restrict__ wsF) {
    int idx = blockIdx.x * blockDim.x + threadIdx.x;
    if (idx < 61440) {
        _Float16 v = (_Float16)0.0f;
        if (idx < 8192) {
            int k = idx >> 8, n = idx & 255;
            if (k < IN_DIM) v = (_Float16)w_ih0[n * IN_DIM + k];
        } else if (idx < 24576) {
            int r = idx - 8192; int k = r >> 8, n = r & 255;
            v = (_Float16)w_hh0[n * H_DIM + k];
        } else if (idx < 40960) {
            int r = idx - 24576; int k = r >> 8, n = r & 255;
            v = (_Float16)w_ih1[n * H_DIM + k];
        } else if (idx < 57344) {
            int r = idx - 40960; int k = r >> 8, n = r & 255;
            v = (_Float16)w_hh1[n * H_DIM + k];
        } else {
            int r = idx - 57344; int k = r >> 5, n = r & 31;
            if (n == 0)       v = (_Float16)w_b0[k];
            else if (n <= KQ) v = (_Float16)w_g[(n - 1) * 128 + k];
        }
        wsH[idx] = v;
    } else if (idx < 61440 + 544) {
        int j = idx - 61440;
        float v;
        if (j < 256)       v = b_ih0[j] + b_hh0[j];
        else if (j < 512) { int n = j - 256; v = b_ih1[n] + b_hh1[n]; }
        else { int m = j - 512; v = (m == 0) ? bb_b0[0] : (m <= KQ ? b_g[m - 1] : 0.0f); }
        wsF[j] = v;
    }
}

__global__ void zero_out_kernel(float* out) {
    if (threadIdx.x == 0 && blockIdx.x == 0) *out = 0.0f;
}

// ---------------- main fused LSTM + heads + CRPS ----------------
__global__ __launch_bounds__(512) void lstm_crps_kernel(
        const float* __restrict__ train, const float* __restrict__ labels,
        const _Float16* __restrict__ wsH, const float* __restrict__ wsF,
        float* __restrict__ out, int Btot) {
    const _Float16* Bih0 = wsH;
    const _Float16* Bhh0 = wsH + 8192;
    const _Float16* Bih1 = wsH + 24576;
    const _Float16* Bhh1 = wsH + 40960;
    const _Float16* Bhd  = wsH + 57344;
    const float* bias0 = wsF;
    const float* bias1 = wsF + 256;
    const float* biasH = wsF + 512;

    // LDS (padded strides chosen to be 16B-aligned and bank-conflict-free)
    __shared__ __attribute__((aligned(16))) _Float16 xA[16 * 40];     // A: x_t    [16][32] pad40
    __shared__ __attribute__((aligned(16))) _Float16 h0A[16 * 72];    // A: h0     [16][64] pad72
    __shared__ __attribute__((aligned(16))) _Float16 h1A[16 * 72];    // A: h1     [16][64] pad72
    __shared__ __attribute__((aligned(16))) _Float16 featsA[16 * 136];// A: feats  [16][128] pad136
    __shared__ float gates[16 * 256];
    __shared__ float headD[16 * 32];
    __shared__ float bArr[16 * KQ];
    __shared__ float knotArr[16 * KQ];
    __shared__ float red[16];
    // head B-fragment register images: [wave 0..1][frag 0..3][lane 0..31][2 x u32x4]
    __shared__ __attribute__((aligned(16))) u32x4 hdStage[2 * 4 * 32 * 2];

    const int tid  = threadIdx.x;
    const int wave = tid >> 5;
    const int lane = tid & 31;
    const int b0row = blockIdx.x * 16;
    const int colT = wave * 16;

    // Resident B fragments for gate GEMMs (live in VGPRs for the whole T loop)
    v16h fih0   = load_b_frag(Bih0, 256, colT, 0);
    v16h fhh0_0 = load_b_frag(Bhh0, 256, colT, 0);
    v16h fhh0_1 = load_b_frag(Bhh0, 256, colT, 32);
    v16h fih1_0 = load_b_frag(Bih1, 256, colT, 0);
    v16h fih1_1 = load_b_frag(Bih1, 256, colT, 32);
    v16h fhh1_0 = load_b_frag(Bhh1, 256, colT, 0);
    v16h fhh1_1 = load_b_frag(Bhh1, 256, colT, 32);

    // Head B fragments -> LDS stage (not kept in registers: avoids VGPR spills)
    float biasHv = 0.0f;
    if (wave < 2) {
        int colH = wave * 16;
#pragma unroll
        for (int f = 0; f < 4; ++f) {
            union { v16h v; u32x4 q[2]; } u;
            u.v = load_b_frag(Bhd, 32, colH, 32 * f);
            u32x4* p = &hdStage[(((wave << 2) | f) * 32 + lane) * 2];
            p[0] = u.q[0];
            p[1] = u.q[1];
        }
        biasHv = biasH[colH + (lane & 15)];
    }
    const float bias0v = bias0[colT + (lane & 15)];
    const float bias1v = bias1[colT + (lane & 15)];

    // zero recurrent state
    for (int i = tid; i < 16 * 72; i += 512) { h0A[i] = (_Float16)0.0f; h1A[i] = (_Float16)0.0f; }
    for (int i = tid; i < 16 * 136; i += 512) featsA[i] = (_Float16)0.0f;

    // cell state in registers: two elements per thread per layer
    float c0a = 0.0f, c0b = 0.0f, c1a = 0.0f, c1b = 0.0f;
    const int e0 = tid, e1 = tid + 512;
    const int ar0 = e0 >> 6, ac0 = e0 & 63;
    const int ar1 = e1 >> 6, ac1 = e1 & 63;
    const int xr = tid >> 5, xc = tid & 31;
    const int taskR = tid / KQ, taskJ = tid - taskR * KQ;   // valid when tid < 320
    float crpsAcc = 0.0f;

    // ---- software pipeline: prefetch x_0 / label_0 with incrementing pointers ----
    const bool xActive = (xc < IN_DIM);
    const float* xPtr = train + (size_t)(b0row + xr) * (T_STEPS * IN_DIM) + xc;
    float xReg = 0.0f;
    if (xActive) xReg = __builtin_nontemporal_load(xPtr);
    const float* labPtr = labels + (size_t)(b0row + (tid & 15)) * T_STEPS;
    float labReg = 0.0f;
    if (tid < 16) labReg = __builtin_nontemporal_load(labPtr);

    __syncthreads();

    for (int t = 0; t < T_STEPS; ++t) {
        // ---- stage 1: x_t (prefetched) -> LDS, then issue prefetch of x_{t+1} ----
        xA[xr * 40 + xc] = (_Float16)xReg;
        xPtr += IN_DIM;
        if (xActive && t + 1 < T_STEPS) xReg = __builtin_nontemporal_load(xPtr);
        __syncthreads();

        // ---- stage 2: layer0 gates = x@Wih0^T + h0@Whh0^T + bias ----
        {
            v8f acc;
#pragma unroll
            for (int r = 0; r < 8; ++r) acc[r] = bias0v;
            acc = wmma_f16(load_a_frag(xA, 40, 0), fih0, acc);
            acc = wmma_f16(load_a_frag(h0A, 72, 0), fhh0_0, acc);
            acc = wmma_f16(load_a_frag(h0A, 72, 32), fhh0_1, acc);
            store_d(gates, 256, colT, acc);
        }
        __syncthreads();

        // ---- stage 3: layer0 activations -> h0A + feats(even) ----
        {
            float gi = gates[ar0 * 256 + ac0],       gf = gates[ar0 * 256 + 64 + ac0];
            float gg = gates[ar0 * 256 + 128 + ac0], go = gates[ar0 * 256 + 192 + ac0];
            float cn = sig_fast(gf) * c0a + sig_fast(gi) * tanh_fast(gg);
            float hn = sig_fast(go) * tanh_fast(cn);
            c0a = cn;
            h0A[ar0 * 72 + ac0] = (_Float16)hn;
            featsA[ar0 * 136 + 2 * ac0] = (_Float16)hn;

            gi = gates[ar1 * 256 + ac1];       gf = gates[ar1 * 256 + 64 + ac1];
            gg = gates[ar1 * 256 + 128 + ac1]; go = gates[ar1 * 256 + 192 + ac1];
            cn = sig_fast(gf) * c0b + sig_fast(gi) * tanh_fast(gg);
            hn = sig_fast(go) * tanh_fast(cn);
            c0b = cn;
            h0A[ar1 * 72 + ac1] = (_Float16)hn;
            featsA[ar1 * 136 + 2 * ac1] = (_Float16)hn;
        }
        __syncthreads();

        // ---- stage 4: layer1 gates = h0'@Wih1^T + h1@Whh1^T + bias ----
        {
            v8f acc;
#pragma unroll
            for (int r = 0; r < 8; ++r) acc[r] = bias1v;
            acc = wmma_f16(load_a_frag(h0A, 72, 0), fih1_0, acc);
            acc = wmma_f16(load_a_frag(h0A, 72, 32), fih1_1, acc);
            acc = wmma_f16(load_a_frag(h1A, 72, 0), fhh1_0, acc);
            acc = wmma_f16(load_a_frag(h1A, 72, 32), fhh1_1, acc);
            store_d(gates, 256, colT, acc);
        }
        __syncthreads();

        // ---- stage 5: layer1 activations -> h1A + feats(odd) ----
        {
            float gi = gates[ar0 * 256 + ac0],       gf = gates[ar0 * 256 + 64 + ac0];
            float gg = gates[ar0 * 256 + 128 + ac0], go = gates[ar0 * 256 + 192 + ac0];
            float cn = sig_fast(gf) * c1a + sig_fast(gi) * tanh_fast(gg);
            float hn = sig_fast(go) * tanh_fast(cn);
            c1a = cn;
            h1A[ar0 * 72 + ac0] = (_Float16)hn;
            featsA[ar0 * 136 + 2 * ac0 + 1] = (_Float16)hn;

            gi = gates[ar1 * 256 + ac1];       gf = gates[ar1 * 256 + 64 + ac1];
            gg = gates[ar1 * 256 + 128 + ac1]; go = gates[ar1 * 256 + 192 + ac1];
            cn = sig_fast(gf) * c1b + sig_fast(gi) * tanh_fast(gg);
            hn = sig_fast(go) * tanh_fast(cn);
            c1b = cn;
            h1A[ar1 * 72 + ac1] = (_Float16)hn;
            featsA[ar1 * 136 + 2 * ac1 + 1] = (_Float16)hn;
        }
        __syncthreads();

        // ---- stage 6: heads (waves 0,1): [beta0 | gamma0..19] = feats @ Wh^T ----
        if (wave < 2) {
            const u32x4* st = &hdStage[((wave << 2) * 32 + lane) * 2];
            v8f acc;
#pragma unroll
            for (int r = 0; r < 8; ++r) acc[r] = biasHv;
#pragma unroll
            for (int f = 0; f < 4; ++f) {
                union { v16h v; u32x4 q[2]; } u;
                u.q[0] = st[f * 64 + 0];
                u.q[1] = st[f * 64 + 1];
                acc = wmma_f16(load_a_frag(featsA, 136, 32 * f), u.v, acc);
            }
            store_d(headD, 32, wave * 16, acc);
        }
        __syncthreads();

        // ---- stage 7: softplus on head outputs (21 cols) ----
        {
            int row = tid >> 5, col = tid & 31;
            if (col <= KQ) headD[row * 32 + col] = softplus_fast(headD[row * 32 + col]);
        }
        __syncthreads();

        // ---- stage 8: spline coefficients b_j (j=19 slot fixed up locally in stage 10) ----
        if (tid < 16 * KQ) {
            int row = taskR, j = taskJ;
            float bet0 = headD[row * 32 + 0];
            float gj = headD[row * 32 + 1 + j];
            float bpj = (j == 0) ? bet0 : headD[row * 32 + j];
            float rj = (gj - bpj) * (0.5f * (float)KQ);   // /(2*sigma), sigma=1/K
            float rjm1 = 0.0f;
            if (j > 0) {
                float gjm  = headD[row * 32 + j];
                float bpjm = (j == 1) ? bet0 : headD[row * 32 + j - 1];
                rjm1 = (gjm - bpjm) * (0.5f * (float)KQ);
            }
            bArr[row * KQ + j] = rj - rjm1;
        }
        __syncthreads();

        // ---- stage 9: knot values knots_i = b0*ksi_i + sum_{j<i} b_j*(ksi_i-ksi_j)^2 ----
        // (only j<=18 ever contributes: b19 not needed here)
        if (tid < 16 * KQ) {
            int row = taskR, i = taskJ;
            float b0v = headD[row * 32 + 0];
            float kn = b0v * ((float)i * (1.0f / (float)KQ));
            for (int j = 0; j < i; ++j) {
                float d = (float)(i - j) * (1.0f / (float)KQ);
                kn += bArr[row * KQ + j] * d * d;
            }
            knotArr[row * KQ + i] = kn;
        }
        __syncthreads();

        // ---- stage 10: per-row CRPS (prefetches next label at the end) ----
        if (tid < 16) {
            int row = tid;
            float lab = labReg;
            float b0v = headD[row * 32 + 0];
            float Asum = 0.0f, Bsum = 0.0f, Csum = 0.0f, btSum = 0.0f;
            float bestAd = 3.4e38f, ksiPick = 0.0f;
#pragma unroll
            for (int j = 0; j < KQ - 1; ++j) {
                float bj = bArr[row * KQ + j];
                float kj = knotArr[row * KQ + j];
                float ksij = (float)j * (1.0f / (float)KQ);
                float diff = lab - kj;
                float al = (diff > 0.0f) ? 1.0f : 0.0f;
                btSum += bj;
                Asum += al * bj;
                Bsum += al * bj * ksij;
                Csum += al * bj * ksij * ksij;
                float ad = fabsf(diff);
                if (ad < bestAd) { bestAd = ad; ksiPick = ksij; }
            }
            // j = 19: b19 = gamma19 - sum_{j<19} b_j (local fix-up, no extra barrier)
            float b19 = headD[row * 32 + KQ] - btSum;
            {
                float kj = knotArr[row * KQ + (KQ - 1)];
                float ksij = (float)(KQ - 1) * (1.0f / (float)KQ);
                float diff = lab - kj;
                float al = (diff > 0.0f) ? 1.0f : 0.0f;
                Asum += al * b19;
                Bsum += al * b19 * ksij;
                Csum += al * b19 * ksij * ksij;
                float ad = fabsf(diff);
                if (ad < bestAd) { bestAd = ad; ksiPick = ksij; }
            }
            float Bq = b0v - 2.0f * Bsum;
            float Cq = -lab + Csum;
            float disc = Bq * Bq - 4.0f * Asum * Cq;
            float Asafe = (Asum != 0.0f) ? Asum : 1.0f;
            float Bsafe = (Bq != 0.0f) ? Bq : 1.0f;
            float quad = (-Bq + sqrtf(fmaxf(disc, 0.0f))) / (2.0f * Asafe);
            float alpha = (Asum != 0.0f && disc >= 0.0f) ? quad
                          : ((Asum == 0.0f) ? (-Cq / Bsafe) : ksiPick);
            float c1v = lab * (2.0f * alpha - 1.0f);
            float c2v = b0v * (1.0f / 3.0f - alpha * alpha);
            float c3v = 0.0f, c4v = 0.0f;
#pragma unroll
            for (int j = 0; j < KQ; ++j) {
                float bj = (j == KQ - 1) ? b19 : bArr[row * KQ + j];
                float ksij = (float)j * (1.0f / (float)KQ);
                float om = 1.0f - ksij, om2 = om * om;
                c3v += bj * om2 * om2 * (1.0f / 6.0f);
                float al = ((lab - knotArr[row * KQ + j]) > 0.0f) ? 1.0f : 0.0f;
                float d = alpha - ksij;
                c4v += al * (2.0f / 3.0f) * bj * d * d * d;
            }
            crpsAcc += c1v + c2v + c3v - c4v;
            ++labPtr;
            if (t + 1 < T_STEPS) labReg = __builtin_nontemporal_load(labPtr);
        }
        // next-iteration's first __syncthreads orders stage 10 vs. reuse of buffers
    }

    if (tid < 16) red[tid] = crpsAcc;
    __syncthreads();
    if (tid == 0) {
        float s = 0.0f;
#pragma unroll
        for (int i = 0; i < 16; ++i) s += red[i];
        atomicAdd(out, s / (float)Btot);   // mean over batch, summed over t
    }
}

// ---------------- launcher ----------------
extern "C" void kernel_launch(void* const* d_in, const int* in_sizes, int n_in,
                              void* d_out, int out_size, void* d_ws, size_t ws_size,
                              hipStream_t stream) {
    const float* train  = (const float*)d_in[0];
    const float* labels = (const float*)d_in[1];
    const float* w_ih0  = (const float*)d_in[2];
    const float* w_hh0  = (const float*)d_in[3];
    const float* b_ih0  = (const float*)d_in[4];
    const float* b_hh0  = (const float*)d_in[5];
    const float* w_ih1  = (const float*)d_in[6];
    const float* w_hh1  = (const float*)d_in[7];
    const float* b_ih1  = (const float*)d_in[8];
    const float* b_hh1  = (const float*)d_in[9];
    const float* w_b0   = (const float*)d_in[10];
    const float* bb_b0  = (const float*)d_in[11];
    const float* w_g    = (const float*)d_in[12];
    const float* b_g    = (const float*)d_in[13];

    int Btot = in_sizes[0] / (T_STEPS * IN_DIM);   // 2048

    _Float16* wsH = (_Float16*)d_ws;
    float* wsF = (float*)((char*)d_ws + 61440 * sizeof(_Float16));
    float* out = (float*)d_out;

    zero_out_kernel<<<1, 1, 0, stream>>>(out);
    prep_kernel<<<(61440 + 544 + 255) / 256, 256, 0, stream>>>(
        w_ih0, w_hh0, b_ih0, b_hh0, w_ih1, w_hh1, b_ih1, b_hh1,
        w_b0, bb_b0, w_g, b_g, wsH, wsF);
    lstm_crps_kernel<<<Btot / 16, 512, 0, stream>>>(train, labels, wsH, wsF, out, Btot);
}
